// MultiHeadGeometricAttention_44384192037443
// MI455X (gfx1250) — compile-verified, matrix operand order checked by reading, not run on hardware
//
#include <hip/hip_runtime.h>

typedef __attribute__((ext_vector_type(16))) _Float16 v16h;
typedef __attribute__((ext_vector_type(8)))  _Float16 v8h;
typedef __attribute__((ext_vector_type(8)))  float    v8f;

#define Bc  2
#define Tc  2048
#define DMc 768
#define Hc  12
#define Dc  64

static constexpr size_t NXH  = (size_t)Bc * Tc * DMc;     // 3,145,728 halves
static constexpr size_t NW   = (size_t)DMc * DMc;         //   589,824
static constexpr size_t NQA  = (size_t)Bc * Hc * Tc * 96; // 4,718,592
static constexpr size_t NVT  = (size_t)Bc * Hc * Dc * Tc; // 3,145,728

static constexpr size_t OFF_XH  = 0;
static constexpr size_t OFF_WQ  = OFF_XH + NXH;
static constexpr size_t OFF_WK  = OFF_WQ + NW;
static constexpr size_t OFF_WV  = OFF_WK + NW;
static constexpr size_t OFF_WO  = OFF_WV + NW;
static constexpr size_t OFF_QA  = OFF_WO + NW;
static constexpr size_t OFF_KA  = OFF_QA + NQA;
static constexpr size_t OFF_VT  = OFF_KA + NQA;
static constexpr size_t OFF_CTX = OFF_VT + NVT;

// ---------------------------------------------------------------------------
// Fragment helpers.
// A-fragment mapping assumed: row = lane&15; halves h: k = (h>>3)*16 +
// (lane>>4)*8 + (h&7). B-fragment uses the mirrored mapping with col=lane&15,
// loaded from a row-major [N,K] matrix (i.e. B[k][n] = M[n][k]). Any internal
// k-permutation cancels between A and B, so the contraction is correct as
// long as both use the same scheme and the documented C layout holds.
// ---------------------------------------------------------------------------
__device__ __forceinline__ v16h load_frag_ptr(const _Float16* p) {
  v8h lo = *(const v8h*)p;
  v8h hi = *(const v8h*)(p + 16);
  v16h f;
#pragma unroll
  for (int i = 0; i < 8; ++i) { f[i] = lo[i]; f[i + 8] = hi[i]; }
  return f;
}

__device__ __forceinline__ v16h load_frag_rm(const _Float16* base, int ld,
                                             int r0, int k0, int lane) {
  return load_frag_ptr(base + (size_t)(r0 + (lane & 15)) * ld + k0 +
                       ((lane >> 4) << 3));
}

__device__ __forceinline__ v8f wmma_f16(v16h a, v16h b, v8f c) {
  return __builtin_amdgcn_wmma_f32_16x16x32_f16(false, a, false, b, (short)0,
                                                c, false, false);
}

// ---------------------------------------------------------------------------
// 0) fp32 -> f16 conversion of x and the four weight matrices
// ---------------------------------------------------------------------------
__global__ void cvt_kernel(const float* x, const float* wq, const float* wk,
                           const float* wv, const float* wo, _Float16* ws) {
  size_t i = (size_t)blockIdx.x * blockDim.x + threadIdx.x;
  if (i < NXH) { ws[OFF_XH + i] = (_Float16)x[i]; return; }
  size_t j = i - NXH;
  if (j < NW) { ws[OFF_WQ + j] = (_Float16)wq[j]; return; }
  j -= NW;
  if (j < NW) { ws[OFF_WK + j] = (_Float16)wk[j]; return; }
  j -= NW;
  if (j < NW) { ws[OFF_WV + j] = (_Float16)wv[j]; return; }
  j -= NW;
  if (j < NW) { ws[OFF_WO + j] = (_Float16)wo[j]; return; }
}

// ---------------------------------------------------------------------------
// 1) QKV projection: [4096,768] x [768,768]^T, one wave per 32x64 tile
//    (two M sub-tiles share each B fragment -> ~22 FLOP/byte of L2 traffic,
//    and two independent WMMAs per B load for latency overlap).
//    mode (blockIdx.z): 0=Q (scaled 1/8 into Qa), 1=K (into Ka), 2=V (into Vt,
//    stored transposed [d][t] per (b,h)).
// ---------------------------------------------------------------------------
__global__ __launch_bounds__(128) void proj_kernel(
    const _Float16* xh, const _Float16* wq, const _Float16* wk,
    const _Float16* wv, _Float16* Qa, _Float16* Ka, _Float16* Vt) {
  int lane  = threadIdx.x & 31;
  int wavei = threadIdx.x >> 5;
  int m0 = (blockIdx.y * 4 + wavei) * 32;   // 128 M-tiles of 32 rows
  int n0 = blockIdx.x * 64;
  int mode = blockIdx.z;
  const _Float16* W = (mode == 0) ? wq : (mode == 1) ? wk : wv;

  v8f z = {};
  v8f acc[2][4] = {{z, z, z, z}, {z, z, z, z}};
  for (int k0 = 0; k0 < DMc; k0 += 32) {
    v16h a0 = load_frag_rm(xh, DMc, m0, k0, lane);
    v16h a1 = load_frag_rm(xh, DMc, m0 + 16, k0, lane);
#pragma unroll
    for (int ti = 0; ti < 4; ++ti) {
      v16h b = load_frag_rm(W, DMc, n0 + ti * 16, k0, lane);
      acc[0][ti] = wmma_f16(a0, b, acc[0][ti]);
      acc[1][ti] = wmma_f16(a1, b, acc[1][ti]);
    }
  }

  int ncol = lane & 15;
  int rb = (lane >> 4) << 3;
#pragma unroll
  for (int mi = 0; mi < 2; ++mi) {
#pragma unroll
    for (int ti = 0; ti < 4; ++ti) {
#pragma unroll
      for (int r = 0; r < 8; ++r) {
        int mg = m0 + mi * 16 + rb + r;
        int bb = mg >> 11;            // T = 2048
        int t  = mg & (Tc - 1);
        int ng = n0 + ti * 16 + ncol;
        int h = ng >> 6, d = ng & 63;
        float vv = acc[mi][ti][r];
        if (mode == 0)
          Qa[(((size_t)(bb * Hc + h)) * Tc + t) * 96 + d] =
              (_Float16)(vv * 0.125f);
        else if (mode == 1)
          Ka[(((size_t)(bb * Hc + h)) * Tc + t) * 96 + d] = (_Float16)vv;
        else
          Vt[(((size_t)(bb * Hc + h)) * Dc + d) * Tc + t] = (_Float16)vv;
      }
    }
  }
}

// ---------------------------------------------------------------------------
// 2) Geometric codebook weights -> augmented dims [64:96) of Qa / Ka.
//    One thread per (b,h,t). Qa holds q*0.125, so recover q with *8.
// ---------------------------------------------------------------------------
__global__ void geo_kernel(const float* Wqg, const float* Wkg, const float* cb,
                           const float* log_temps, const float* geo_scale,
                           _Float16* Qa, _Float16* Ka) {
  int idx = blockIdx.x * blockDim.x + threadIdx.x;
  if (idx >= Bc * Hc * Tc) return;
  int h = (idx / Tc) % Hc;
  float temp = __expf(log_temps[h]);
  temp = fminf(fmaxf(temp, 0.01f), 5.0f);
  float inv_t = 1.0f / temp;
  float gs = geo_scale[0];
  const float* cbh = cb + h * 24;

#pragma unroll
  for (int side = 0; side < 2; ++side) {
    _Float16* base = (side == 0 ? Qa : Ka) + (size_t)idx * 96;
    const float* Wg = (side == 0) ? Wqg : Wkg;
    float mul = (side == 0) ? 8.0f : 1.0f;
    float g0 = 0.f, g1 = 0.f, g2 = 0.f;
    for (int d2 = 0; d2 < 64; ++d2) {
      float qv = mul * (float)base[d2];
      g0 += qv * Wg[d2];
      g1 += qv * Wg[64 + d2];
      g2 += qv * Wg[128 + d2];
    }
    float lg[8];
    float mx = -1e30f;
#pragma unroll
    for (int j = 0; j < 8; ++j) {
      float e0 = g0 - cbh[j * 3 + 0];
      float e1 = g1 - cbh[j * 3 + 1];
      float e2 = g2 - cbh[j * 3 + 2];
      lg[j] = -(e0 * e0 + e1 * e1 + e2 * e2) * inv_t;
      mx = fmaxf(mx, lg[j]);
    }
    float se = 0.f;
#pragma unroll
    for (int j = 0; j < 8; ++j) { lg[j] = __expf(lg[j] - mx); se += lg[j]; }
    float sc = ((side == 0) ? gs : 1.0f) / se;
    _Float16* dst = base + 64;
#pragma unroll
    for (int j = 0; j < 8; ++j) dst[j] = (_Float16)(lg[j] * sc);
#pragma unroll
    for (int j = 8; j < 32; ++j) dst[j] = (_Float16)0.0f;  // zero pad to 96
  }
}

// ---------------------------------------------------------------------------
// 3) Flash attention over augmented 96-dim vectors. One wave per 16 query
//    rows; 32 key columns per step (6 score WMMAs + 4 P*V WMMAs).
// ---------------------------------------------------------------------------
__global__ __launch_bounds__(32) void attn_kernel(const _Float16* Qa,
                                                  const _Float16* Ka,
                                                  const _Float16* Vt,
                                                  _Float16* ctx) {
  __shared__ __align__(16) _Float16 Plds[16 * 32];
  int lane = threadIdx.x;
  int mt = blockIdx.x, h = blockIdx.y, b = blockIdx.z;
  const _Float16* Qbh = Qa + ((size_t)(b * Hc + h)) * Tc * 96;
  const _Float16* Kbh = Ka + ((size_t)(b * Hc + h)) * Tc * 96;
  const _Float16* Vbh = Vt + ((size_t)(b * Hc + h)) * Dc * Tc;
  int m0 = mt * 16;

  v16h a0 = load_frag_rm(Qbh, 96, m0, 0, lane);
  v16h a1 = load_frag_rm(Qbh, 96, m0, 32, lane);
  v16h a2 = load_frag_rm(Qbh, 96, m0, 64, lane);

  v8f z = {};
  v8f o0 = z, o1 = z, o2 = z, o3 = z;
  float m_i[8], l_i[8];
#pragma unroll
  for (int r = 0; r < 8; ++r) { m_i[r] = -1e30f; l_i[r] = 0.f; }

  int ncol = lane & 15;
  int rb = (lane >> 4) << 3;
  int send = m0 + 16;  // exclusive causal column bound for this row tile

  for (int s0 = 0; s0 < send; s0 += 32) {
    __builtin_prefetch(Kbh + (size_t)(s0 + 32) * 96, 0, 1);
    v8f sA = z, sB = z;
    {
      v16h b0 = load_frag_rm(Kbh, 96, s0, 0, lane);
      sA = wmma_f16(a0, b0, sA);
      v16h b1 = load_frag_rm(Kbh, 96, s0, 32, lane);
      sA = wmma_f16(a1, b1, sA);
      v16h b2 = load_frag_rm(Kbh, 96, s0, 64, lane);
      sA = wmma_f16(a2, b2, sA);
      v16h c0 = load_frag_rm(Kbh, 96, s0 + 16, 0, lane);
      sB = wmma_f16(a0, c0, sB);
      v16h c1 = load_frag_rm(Kbh, 96, s0 + 16, 32, lane);
      sB = wmma_f16(a1, c1, sB);
      v16h c2 = load_frag_rm(Kbh, 96, s0 + 16, 64, lane);
      sB = wmma_f16(a2, c2, sB);
    }
    // causal mask
#pragma unroll
    for (int r = 0; r < 8; ++r) {
      int row = m0 + rb + r;
      if (s0 + ncol > row) sA[r] = -1e30f;
      if (s0 + 16 + ncol > row) sB[r] = -1e30f;
    }
    // online softmax: row max across the 16 lanes sharing each row
    float rmax[8];
#pragma unroll
    for (int r = 0; r < 8; ++r) rmax[r] = fmaxf(sA[r], sB[r]);
#pragma unroll
    for (int r = 0; r < 8; ++r) {
#pragma unroll
      for (int off = 1; off < 16; off <<= 1)
        rmax[r] = fmaxf(rmax[r], __shfl_xor(rmax[r], off, 32));
    }
    float cf[8], p0[8], p1[8], rs[8];
#pragma unroll
    for (int r = 0; r < 8; ++r) {
      float mn = fmaxf(m_i[r], rmax[r]);
      cf[r] = __expf(m_i[r] - mn);
      m_i[r] = mn;
      p0[r] = __expf(sA[r] - mn);
      p1[r] = __expf(sB[r] - mn);
      rs[r] = p0[r] + p1[r];
    }
#pragma unroll
    for (int r = 0; r < 8; ++r) {
#pragma unroll
      for (int off = 1; off < 16; off <<= 1)
        rs[r] += __shfl_xor(rs[r], off, 32);
      l_i[r] = l_i[r] * cf[r] + rs[r];
    }
#pragma unroll
    for (int r = 0; r < 8; ++r) {
      o0[r] *= cf[r]; o1[r] *= cf[r]; o2[r] *= cf[r]; o3[r] *= cf[r];
    }
    // C-layout -> A-layout relayout of P through LDS
#pragma unroll
    for (int r = 0; r < 8; ++r) {
      Plds[(rb + r) * 32 + ncol] = (_Float16)p0[r];
      Plds[(rb + r) * 32 + 16 + ncol] = (_Float16)p1[r];
    }
    asm volatile("s_wait_dscnt 0" ::: "memory");
    v16h pf = load_frag_rm(Plds, 32, 0, 0, lane);
    v16h v0 = load_frag_rm(Vbh, Tc, 0, s0, lane);
    o0 = wmma_f16(pf, v0, o0);
    v16h v1 = load_frag_rm(Vbh, Tc, 16, s0, lane);
    o1 = wmma_f16(pf, v1, o1);
    v16h v2 = load_frag_rm(Vbh, Tc, 32, s0, lane);
    o2 = wmma_f16(pf, v2, o2);
    v16h v3 = load_frag_rm(Vbh, Tc, 48, s0, lane);
    o3 = wmma_f16(pf, v3, o3);
  }

  // normalize and store ctx[b, t, h*64 + d] (f16)
  _Float16* cbase = ctx + ((size_t)b * Tc + m0) * DMc + h * 64;
#pragma unroll
  for (int r = 0; r < 8; ++r) {
    float inv = 1.0f / l_i[r];
    size_t t = (size_t)(rb + r);
    cbase[t * DMc + 0 + ncol]  = (_Float16)(o0[r] * inv);
    cbase[t * DMc + 16 + ncol] = (_Float16)(o1[r] * inv);
    cbase[t * DMc + 32 + ncol] = (_Float16)(o2[r] * inv);
    cbase[t * DMc + 48 + ncol] = (_Float16)(o3[r] * inv);
  }
}

// ---------------------------------------------------------------------------
// 4) Output projection: out = ctx @ Wo^T, fp32 result straight to d_out.
//    Same 32x64 wave tile as proj_kernel.
// ---------------------------------------------------------------------------
__global__ __launch_bounds__(128) void oproj_kernel(const _Float16* ctx,
                                                    const _Float16* wo,
                                                    float* out) {
  int lane  = threadIdx.x & 31;
  int wavei = threadIdx.x >> 5;
  int m0 = (blockIdx.y * 4 + wavei) * 32;
  int n0 = blockIdx.x * 64;

  v8f z = {};
  v8f acc[2][4] = {{z, z, z, z}, {z, z, z, z}};
  for (int k0 = 0; k0 < DMc; k0 += 32) {
    v16h a0 = load_frag_rm(ctx, DMc, m0, k0, lane);
    v16h a1 = load_frag_rm(ctx, DMc, m0 + 16, k0, lane);
#pragma unroll
    for (int ti = 0; ti < 4; ++ti) {
      v16h b = load_frag_rm(wo, DMc, n0 + ti * 16, k0, lane);
      acc[0][ti] = wmma_f16(a0, b, acc[0][ti]);
      acc[1][ti] = wmma_f16(a1, b, acc[1][ti]);
    }
  }
  int ncol = lane & 15;
  int rb = (lane >> 4) << 3;
#pragma unroll
  for (int mi = 0; mi < 2; ++mi) {
#pragma unroll
    for (int ti = 0; ti < 4; ++ti) {
#pragma unroll
      for (int r = 0; r < 8; ++r) {
        size_t mg = (size_t)(m0 + mi * 16 + rb + r);
        int ng = n0 + ti * 16 + ncol;
        out[mg * DMc + ng] = acc[mi][ti][r];
      }
    }
  }
}

// ---------------------------------------------------------------------------
extern "C" void kernel_launch(void* const* d_in, const int* in_sizes, int n_in,
                              void* d_out, int out_size, void* d_ws,
                              size_t ws_size, hipStream_t stream) {
  (void)in_sizes; (void)n_in; (void)out_size; (void)ws_size;
  const float* x         = (const float*)d_in[0];
  const float* Wq        = (const float*)d_in[1];
  const float* Wk        = (const float*)d_in[2];
  const float* Wv        = (const float*)d_in[3];
  const float* Wo        = (const float*)d_in[4];
  const float* Wqg       = (const float*)d_in[5];
  const float* Wkg       = (const float*)d_in[6];
  const float* codebooks = (const float*)d_in[7];
  const float* log_temps = (const float*)d_in[8];
  const float* geo_scale = (const float*)d_in[9];

  _Float16* ws  = (_Float16*)d_ws;
  _Float16* xh  = ws + OFF_XH;
  _Float16* wqh = ws + OFF_WQ;
  _Float16* wkh = ws + OFF_WK;
  _Float16* wvh = ws + OFF_WV;
  _Float16* woh = ws + OFF_WO;
  _Float16* Qa  = ws + OFF_QA;
  _Float16* Ka  = ws + OFF_KA;
  _Float16* Vt  = ws + OFF_VT;
  _Float16* ctx = ws + OFF_CTX;
  float* out = (float*)d_out;

  // 0) convert inputs to f16
  {
    size_t n = NXH + 4 * NW;
    int grid = (int)((n + 255) / 256);
    cvt_kernel<<<grid, 256, 0, stream>>>(x, Wq, Wk, Wv, Wo, ws);
  }
  // 1) Q/K/V projections (z = 0/1/2); 128 M-tiles of 32 rows, 12 N-strips
  proj_kernel<<<dim3(12, 32, 3), 128, 0, stream>>>(xh, wqh, wkh, wvh, Qa, Ka,
                                                   Vt);
  // 2) geometric codebook softmax weights into the augmented dims
  geo_kernel<<<(Bc * Hc * Tc + 255) / 256, 256, 0, stream>>>(
      Wqg, Wkg, codebooks, log_temps, geo_scale, Qa, Ka);
  // 3) flash attention (1 wave per 16 rows)
  attn_kernel<<<dim3(Tc / 16, Hc, Bc), 32, 0, stream>>>(Qa, Ka, Vt, ctx);
  // 4) output projection -> fp32 out
  oproj_kernel<<<dim3(12, 32, 1), 128, 0, stream>>>(ctx, woh, out);
}
